// TGATNodeBaseline_1065151889636
// MI455X (gfx1250) — compile-verified, matrix operand order checked by reading, not run on hardware
//
#include <hip/hip_runtime.h>
#include <hip/hip_bf16.h>
#include <math.h>

#define N_NODES 50000
#define N_EDGES 400000
#define NODE_IN 165
#define HID     64
#define HEADS   4
#define HC      256   // HEADS*HID

typedef __attribute__((ext_vector_type(16))) __bf16 v16bf;
typedef __attribute__((ext_vector_type(8)))  float  v8f;

// K index inside a 16x32 bf16 WMMA fragment (ISA 7.12.2, 16-bit A 16x32)
__device__ __forceinline__ int kmap(int e, int half) {
    return e + half * 8 + ((e >= 8) ? 8 : 0);
}

// order-preserving float <-> uint mapping for atomicMax on signed floats
__device__ __forceinline__ unsigned int f2u_ord(float f) {
    unsigned int b = __float_as_uint(f);
    return (b & 0x80000000u) ? ~b : (b | 0x80000000u);
}
__device__ __forceinline__ float u2f_ord(unsigned int u) {
    unsigned int b = (u & 0x80000000u) ? (u & 0x7FFFFFFFu) : ~u;
    return __uint_as_float(b);
}

// vectorized A fragment: two contiguous 8-float groups -> 4x global_load_b128
__device__ __forceinline__ v16bf load_afrag_vec(const float* __restrict__ arow,
                                                int k0, int half) {
    const float4* p0 = reinterpret_cast<const float4*>(arow + k0 + 8 * half);
    const float4* p1 = reinterpret_cast<const float4*>(arow + k0 + 16 + 8 * half);
    float4 a0 = p0[0], a1 = p0[1], a2 = p1[0], a3 = p1[1];
    v16bf f;
    f[0]  = (__bf16)a0.x; f[1]  = (__bf16)a0.y; f[2]  = (__bf16)a0.z; f[3]  = (__bf16)a0.w;
    f[4]  = (__bf16)a1.x; f[5]  = (__bf16)a1.y; f[6]  = (__bf16)a1.z; f[7]  = (__bf16)a1.w;
    f[8]  = (__bf16)a2.x; f[9]  = (__bf16)a2.y; f[10] = (__bf16)a2.z; f[11] = (__bf16)a2.w;
    f[12] = (__bf16)a3.x; f[13] = (__bf16)a3.y; f[14] = (__bf16)a3.z; f[15] = (__bf16)a3.w;
    return f;
}

// ---------------------------------------------------------------------------
// Weight pack: W[K, N] fp32 row-major -> per-lane WMMA B fragments, bf16.
// ---------------------------------------------------------------------------
__global__ void pack_w(const float* __restrict__ W, __bf16* __restrict__ out,
                       int K, int NT) {
    const int N  = NT * 16;
    const int KC = (K + 31) / 32;
    int idx = blockIdx.x * blockDim.x + threadIdx.x;   // (kc*NT + nt)*32 + lane
    if (idx >= KC * NT * 32) return;
    int lane = idx & 31;
    int t    = idx >> 5;
    int nt   = t % NT;
    int kc   = t / NT;
    int half = lane >> 4, r = lane & 15;
    int col  = nt * 16 + r;
    v16bf frag;
#pragma unroll
    for (int e = 0; e < 16; ++e) {
        int kk  = kc * 32 + kmap(e, half);
        float v = (kk < K) ? W[(long)kk * N + col] : 0.0f;
        frag[e] = (__bf16)v;
    }
    *reinterpret_cast<v16bf*>(out + (size_t)idx * 16) = frag;
}

// ---------------------------------------------------------------------------
// q/k/v GEMM: C[M, 256] = A[M,K] @ W + bias.
// Per wave: 2 M-tiles x 8 N-tiles (gridDim.y = 2 n-groups of 8).
// B fragments double-buffered; each feeds 2 WMMAs. Requires K % 32 == 0,
// K % 8 == 0 row alignment, M % 16 == 0.
// ---------------------------------------------------------------------------
__global__ __launch_bounds__(256)
void gemm_qkv(const float* __restrict__ A, const __bf16* __restrict__ Wpk,
              const float* __restrict__ bias, float* __restrict__ C,
              int M, int K) {
    const int NT     = 8;
    const int NTILES = HC / 16;              // 16 n-tiles total
    const int wave   = threadIdx.x >> 5;
    const int lane   = threadIdx.x & 31;
    const int half   = lane >> 4;
    const int r      = lane & 15;
    const int mtiles = M >> 4;
    const int mpairs = (mtiles + 1) >> 1;
    const int pair   = blockIdx.x * 8 + wave;
    if (pair >= mpairs) return;              // whole-wave exit
    const int  g    = blockIdx.y;            // n-group (0 or 1)
    const int  mt0  = pair * 2;
    const bool has2 = (mt0 + 1) < mtiles;
    const int  row0 = mt0 * 16 + r;
    const int  row1 = has2 ? (row0 + 16) : row0;   // alias tile0 when absent

    const float* __restrict__ arow0 = A + (size_t)row0 * K;
    const float* __restrict__ arow1 = A + (size_t)row1 * K;
    const v16bf* __restrict__ bfr   = reinterpret_cast<const v16bf*>(Wpk);
    const int KC = K >> 5;

    v8f acc[2][NT] = {};

    for (int kc = 0; kc < KC; ++kc) {
        const int k0 = kc * 32;
        if (kc + 1 < KC) {                   // pull next A chunk toward the caches
            __builtin_prefetch(arow0 + k0 + 32 + 8 * half, 0, 3);
            __builtin_prefetch(arow1 + k0 + 32 + 8 * half, 0, 3);
        }
        v16bf a0 = load_afrag_vec(arow0, k0, half);
        v16bf a1 = load_afrag_vec(arow1, k0, half);

        const size_t base = (size_t)kc * NTILES + g * NT;
        v16bf bcur = bfr[base * 32 + lane];
#pragma unroll
        for (int nt = 0; nt < NT; ++nt) {
            size_t nidx = (nt + 1 < NT)
                              ? (base + nt + 1)
                              : ((kc + 1 < KC) ? ((size_t)(kc + 1) * NTILES + g * NT)
                                               : base);
            v16bf bnext = bfr[nidx * 32 + lane];          // in flight during WMMAs
            acc[0][nt] = __builtin_amdgcn_wmma_f32_16x16x32_bf16(
                false, a0, false, bcur, (short)0, acc[0][nt], false, false);
            acc[1][nt] = __builtin_amdgcn_wmma_f32_16x16x32_bf16(
                false, a1, false, bcur, (short)0, acc[1][nt], false, false);
            bcur = bnext;
        }
    }

#pragma unroll
    for (int nt = 0; nt < NT; ++nt) {
        int   col = (g * NT + nt) * 16 + r;
        float b   = bias[col];
#pragma unroll
        for (int i = 0; i < 8; ++i) {
            int rr = mt0 * 16 + i + 8 * half;
            C[(size_t)rr * HC + col] = acc[0][nt][i] + b;
        }
        if (has2) {
#pragma unroll
            for (int i = 0; i < 8; ++i) {
                int rr = mt0 * 16 + 16 + i + 8 * half;
                C[(size_t)rr * HC + col] = acc[1][nt][i] + b;
            }
        }
    }
}

// ---------------------------------------------------------------------------
// Projection GEMM: h0[M,64] = relu(x[M,165] @ Wp + bp). Scalar A loads with
// branchless clamp+select tail (K=165). 8 waves/block, 1 m-tile each.
// ---------------------------------------------------------------------------
__global__ __launch_bounds__(256)
void gemm_proj(const float* __restrict__ A, const __bf16* __restrict__ Wpk,
               const float* __restrict__ bias, float* __restrict__ C,
               int M, int K) {
    const int NT    = 4;
    const int wave  = threadIdx.x >> 5;
    const int lane  = threadIdx.x & 31;
    const int half  = lane >> 4;
    const int r     = lane & 15;
    const int mtile = blockIdx.x * 8 + wave;
    if (mtile >= (M >> 4)) return;
    const int m0  = mtile * 16;
    const int row = m0 + r;
    const int KC  = (K + 31) / 32;

    const float* __restrict__ arow = A + (size_t)row * K;
    const v16bf* __restrict__ bfr  = reinterpret_cast<const v16bf*>(Wpk);

    v8f acc[NT] = {};

    for (int kc = 0; kc < KC; ++kc) {
        const int k0 = kc * 32;
        v16bf afrag;
#pragma unroll
        for (int e = 0; e < 16; ++e) {
            int kk  = k0 + kmap(e, half);
            int kcl = (kk < K) ? kk : (K - 1);
            float v = arow[kcl];
            v = (kk < K) ? v : 0.0f;
            afrag[e] = (__bf16)v;
        }
#pragma unroll
        for (int nt = 0; nt < NT; ++nt) {
            v16bf bfrag = bfr[((size_t)kc * NT + nt) * 32 + lane];
            acc[nt] = __builtin_amdgcn_wmma_f32_16x16x32_bf16(
                false, afrag, false, bfrag, (short)0, acc[nt], false, false);
        }
    }

#pragma unroll
    for (int nt = 0; nt < NT; ++nt) {
        int   col = nt * 16 + r;
        float b   = bias[col];
#pragma unroll
        for (int i = 0; i < 8; ++i) {
            int rr  = m0 + i + 8 * half;
            float v = fmaxf(acc[nt][i] + b, 0.0f);
            C[(size_t)rr * HID + col] = v;
        }
    }
}

// ---------------------------------------------------------------------------
__global__ void attn_init(unsigned int* __restrict__ amax,
                          float* __restrict__ denom,
                          float* __restrict__ agg) {
    int idx = blockIdx.x * blockDim.x + threadIdx.x;
    if (idx < N_NODES * HC) agg[idx] = 0.0f;
    if (idx < N_NODES * HEADS) { amax[idx] = 0u; denom[idx] = 0.0f; }
}

// Pass A: alpha[e,h] = (q[dst]·(k[src]+t_enc)) / sqrt(HID); atomicMax per dst.
__global__ __launch_bounds__(256)
void attn_scores(const float* __restrict__ q, const float* __restrict__ k,
                 const int* __restrict__ ei, const float* __restrict__ ea,
                 const float* __restrict__ Wt, const float* __restrict__ bt,
                 float* __restrict__ alpha, unsigned int* __restrict__ amax) {
    const float c_div[8] = {
        1.0f, 0.31622776601683794f, 0.1f, 0.031622776601683791f,
        0.01f, 0.0031622776601683794f, 0.001f, 0.00031622776601683794f };
    int wid  = (int)((blockIdx.x * (long)blockDim.x + threadIdx.x) >> 5);
    int lane = threadIdx.x & 31;
    if (wid >= N_EDGES * HEADS) return;
    int e = wid >> 2, h = wid & 3;
    int src = ei[e], dst = ei[N_EDGES + e];

    float t = ea[2 * e + 1];
    float te[16];
#pragma unroll
    for (int j = 0; j < 8; ++j) {
        float ang = t * c_div[j];
        te[2 * j]     = __sinf(ang);
        te[2 * j + 1] = __cosf(ang);
    }

    float p = 0.0f;
#pragma unroll
    for (int s = 0; s < 2; ++s) {
        int   col  = h * HID + lane + 32 * s;
        float tenc = bt[col];
#pragma unroll
        for (int d = 0; d < 16; ++d) tenc += te[d] * Wt[d * HC + col];
        float kj = k[(size_t)src * HC + col] + tenc;
        float qi = q[(size_t)dst * HC + col];
        p += qi * kj;
    }
#pragma unroll
    for (int off = 16; off >= 1; off >>= 1) p += __shfl_xor(p, off, 32);

    if (lane == 0) {
        float a = p * 0.125f;   // 1/sqrt(64)
        alpha[wid] = a;
        atomicMax(&amax[dst * HEADS + h], f2u_ord(a));
    }
}

// Pass B: ex = exp(alpha - amax[dst]); denom[dst] += ex
__global__ void attn_norm(float* __restrict__ alpha, const int* __restrict__ ei,
                          const unsigned int* __restrict__ amax,
                          float* __restrict__ denom) {
    int idx = blockIdx.x * blockDim.x + threadIdx.x;
    if (idx >= N_EDGES * HEADS) return;
    int e = idx >> 2, h = idx & 3;
    int dst = ei[N_EDGES + e];
    float m  = u2f_ord(amax[dst * HEADS + h]);
    float ex = __expf(alpha[idx] - m);
    alpha[idx] = ex;   // reuse buffer as ex
    atomicAdd(&denom[dst * HEADS + h], ex);
}

// Pass C: agg[dst,h,:] += v[src,h,:] * ex/(denom+eps)
__global__ __launch_bounds__(256)
void attn_aggregate(const float* __restrict__ ex, const float* __restrict__ denom,
                    const float* __restrict__ v, const int* __restrict__ ei,
                    float* __restrict__ agg) {
    int wid  = (int)((blockIdx.x * (long)blockDim.x + threadIdx.x) >> 5);
    int lane = threadIdx.x & 31;
    if (wid >= N_EDGES * HEADS) return;
    int e = wid >> 2, h = wid & 3;
    int src = ei[e], dst = ei[N_EDGES + e];
    float w = ex[wid] / (denom[dst * HEADS + h] + 1e-16f);
#pragma unroll
    for (int s = 0; s < 2; ++s) {
        int col = h * HID + lane + 32 * s;
        atomicAdd(&agg[(size_t)dst * HC + col], v[(size_t)src * HC + col] * w);
    }
}

__global__ void relu_inplace(float* __restrict__ x, int n) {
    int i = blockIdx.x * blockDim.x + threadIdx.x;
    if (i < n) x[i] = fmaxf(x[i], 0.0f);
}

// logits[n] = h[n,:] @ Wc + bc ; one wave per node
__global__ __launch_bounds__(256)
void final_linear(const float* __restrict__ h, const float* __restrict__ Wc,
                  const float* __restrict__ bc, float* __restrict__ out) {
    int wid  = (int)((blockIdx.x * (long)blockDim.x + threadIdx.x) >> 5);
    int lane = threadIdx.x & 31;
    if (wid >= N_NODES) return;
    float p = 0.0f;
#pragma unroll
    for (int j = 0; j < 8; ++j) {
        int c = lane + 32 * j;
        p += h[(size_t)wid * HC + c] * Wc[c];
    }
#pragma unroll
    for (int off = 16; off >= 1; off >>= 1) p += __shfl_xor(p, off, 32);
    if (lane == 0) out[wid] = p + bc[0];
}

// ---------------------------------------------------------------------------
extern "C" void kernel_launch(void* const* d_in, const int* in_sizes, int n_in,
                              void* d_out, int out_size, void* d_ws, size_t ws_size,
                              hipStream_t stream) {
    const float* x   = (const float*)d_in[0];
    const int*   ei  = (const int*)  d_in[1];   // harness dtype mapping: integer -> int32
    const float* ea  = (const float*)d_in[2];
    const float* Wp  = (const float*)d_in[3];  const float* bp  = (const float*)d_in[4];
    const float* Wq1 = (const float*)d_in[5];  const float* bq1 = (const float*)d_in[6];
    const float* Wk1 = (const float*)d_in[7];  const float* bk1 = (const float*)d_in[8];
    const float* Wv1 = (const float*)d_in[9];  const float* bv1 = (const float*)d_in[10];
    const float* Wt1 = (const float*)d_in[11]; const float* bt1 = (const float*)d_in[12];
    const float* Wq2 = (const float*)d_in[13]; const float* bq2 = (const float*)d_in[14];
    const float* Wk2 = (const float*)d_in[15]; const float* bk2 = (const float*)d_in[16];
    const float* Wv2 = (const float*)d_in[17]; const float* bv2 = (const float*)d_in[18];
    const float* Wt2 = (const float*)d_in[19]; const float* bt2 = (const float*)d_in[20];
    const float* Wc  = (const float*)d_in[21]; const float* bc  = (const float*)d_in[22];
    float* out = (float*)d_out;

    // ---- fp32 workspace ----
    float* ws = (float*)d_ws;
    size_t off = 0;
    float* h0    = ws + off; off += (size_t)N_NODES * HID;
    float* q     = ws + off; off += (size_t)N_NODES * HC;
    float* kbuf  = ws + off; off += (size_t)N_NODES * HC;
    float* vbuf  = ws + off; off += (size_t)N_NODES * HC;
    float* alpha = ws + off; off += (size_t)N_EDGES * HEADS;
    unsigned int* amax = (unsigned int*)(ws + off); off += (size_t)N_NODES * HEADS;
    float* denom = ws + off; off += (size_t)N_NODES * HEADS;
    float* agg1  = ws + off; off += (size_t)N_NODES * HC;
    float* agg2  = ws + off; off += (size_t)N_NODES * HC;

    // ---- packed bf16 weight fragments ----
    __bf16* wpk = (__bf16*)(ws + off);
    const size_t SZ_P  = (size_t)6 * 4  * 32 * 16;   // proj : KC=6, NT=4
    const size_t SZ_C1 = (size_t)2 * 16 * 32 * 16;   // conv1: KC=2, NTILES=16
    const size_t SZ_C2 = (size_t)8 * 16 * 32 * 16;   // conv2: KC=8, NTILES=16
    __bf16* Wp_pk  = wpk;                 wpk += SZ_P;
    __bf16* Wq1_pk = wpk;                 wpk += SZ_C1;
    __bf16* Wk1_pk = wpk;                 wpk += SZ_C1;
    __bf16* Wv1_pk = wpk;                 wpk += SZ_C1;
    __bf16* Wq2_pk = wpk;                 wpk += SZ_C2;
    __bf16* Wk2_pk = wpk;                 wpk += SZ_C2;
    __bf16* Wv2_pk = wpk;                 wpk += SZ_C2;

    const int mtiles  = N_NODES / 16;                       // 3125 (exact)
    const int mpairs  = (mtiles + 1) / 2;                   // 1563
    const dim3 qkv_grid((mpairs + 7) / 8, 2);               // 8 waves/block, 2 n-groups
    const int proj_bl = (mtiles + 7) / 8;
    const int edge_wave_blocks = (N_EDGES * HEADS) / 8;     // divides evenly
    const int edge_thr_blocks  = (N_EDGES * HEADS + 255) / 256;
    const int node_elem_blocks = (N_NODES * HC + 255) / 256;
    const int node_wave_blocks = (N_NODES + 7) / 8;

    // pack all weight matrices into WMMA fragment order (bf16)
    pack_w<<<(int)((SZ_P  / 16 + 255) / 256), 256, 0, stream>>>(Wp,  Wp_pk,  NODE_IN, 4);
    pack_w<<<(int)((SZ_C1 / 16 + 255) / 256), 256, 0, stream>>>(Wq1, Wq1_pk, HID, 16);
    pack_w<<<(int)((SZ_C1 / 16 + 255) / 256), 256, 0, stream>>>(Wk1, Wk1_pk, HID, 16);
    pack_w<<<(int)((SZ_C1 / 16 + 255) / 256), 256, 0, stream>>>(Wv1, Wv1_pk, HID, 16);
    pack_w<<<(int)((SZ_C2 / 16 + 255) / 256), 256, 0, stream>>>(Wq2, Wq2_pk, HC, 16);
    pack_w<<<(int)((SZ_C2 / 16 + 255) / 256), 256, 0, stream>>>(Wk2, Wk2_pk, HC, 16);
    pack_w<<<(int)((SZ_C2 / 16 + 255) / 256), 256, 0, stream>>>(Wv2, Wv2_pk, HC, 16);

    // 1) h0 = relu(x @ Wp + bp)
    gemm_proj<<<proj_bl, 256, 0, stream>>>(x, Wp_pk, bp, h0, N_NODES, NODE_IN);

    // ---- conv1 ----
    gemm_qkv<<<qkv_grid, 256, 0, stream>>>(h0, Wq1_pk, bq1, q,    N_NODES, HID);
    gemm_qkv<<<qkv_grid, 256, 0, stream>>>(h0, Wk1_pk, bk1, kbuf, N_NODES, HID);
    gemm_qkv<<<qkv_grid, 256, 0, stream>>>(h0, Wv1_pk, bv1, vbuf, N_NODES, HID);
    attn_init<<<node_elem_blocks, 256, 0, stream>>>(amax, denom, agg1);
    attn_scores<<<edge_wave_blocks, 256, 0, stream>>>(q, kbuf, ei, ea, Wt1, bt1, alpha, amax);
    attn_norm<<<edge_thr_blocks, 256, 0, stream>>>(alpha, ei, amax, denom);
    attn_aggregate<<<edge_wave_blocks, 256, 0, stream>>>(alpha, denom, vbuf, ei, agg1);
    relu_inplace<<<node_elem_blocks, 256, 0, stream>>>(agg1, N_NODES * HC);

    // ---- conv2 ----
    gemm_qkv<<<qkv_grid, 256, 0, stream>>>(agg1, Wq2_pk, bq2, q,    N_NODES, HC);
    gemm_qkv<<<qkv_grid, 256, 0, stream>>>(agg1, Wk2_pk, bk2, kbuf, N_NODES, HC);
    gemm_qkv<<<qkv_grid, 256, 0, stream>>>(agg1, Wv2_pk, bv2, vbuf, N_NODES, HC);
    attn_init<<<node_elem_blocks, 256, 0, stream>>>(amax, denom, agg2);
    attn_scores<<<edge_wave_blocks, 256, 0, stream>>>(q, kbuf, ei, ea, Wt2, bt2, alpha, amax);
    attn_norm<<<edge_thr_blocks, 256, 0, stream>>>(alpha, ei, amax, denom);
    attn_aggregate<<<edge_wave_blocks, 256, 0, stream>>>(alpha, denom, vbuf, ei, agg2);
    relu_inplace<<<node_elem_blocks, 256, 0, stream>>>(agg2, N_NODES * HC);

    // final classifier
    final_linear<<<node_wave_blocks, 256, 0, stream>>>(agg2, Wc, bc, out);
}
